// GNN_6116033430077
// MI455X (gfx1250) — matrix operand order repaired and checked by reading, not compiled, not run on hardware
//
#include <hip/hip_runtime.h>
#include <cstddef>

typedef __attribute__((ext_vector_type(16))) __bf16 v16bf;
typedef __attribute__((ext_vector_type(8)))  __bf16 bf16x8;
typedef __attribute__((ext_vector_type(8)))  float  v8f;

#define DD 128

// ---------------------------------------------------------------------------
// Elementwise fp32 -> bf16 (optional ReLU)
// ---------------------------------------------------------------------------
__global__ __launch_bounds__(256) void cvt_bf16_kernel(
    const float* __restrict__ in, __bf16* __restrict__ outB,
    long long n, int doRelu)
{
    long long i = (long long)blockIdx.x * 256 + threadIdx.x;
    if (i >= n) return;
    float v = in[i];
    if (doRelu) v = v > 0.0f ? v : 0.0f;
    outB[i] = (__bf16)v;
}

// ---------------------------------------------------------------------------
// 128x128 weight transpose + bf16 convert:  Wt[n*128+k] = W[k*128+n]
// ---------------------------------------------------------------------------
__global__ __launch_bounds__(256) void wtrans_kernel(
    const float* __restrict__ W, __bf16* __restrict__ Wt)
{
    int idx = blockIdx.x * 256 + threadIdx.x;   // 0..16383
    int k = idx >> 7, n = idx & 127;
    Wt[n * DD + k] = (__bf16)W[k * DD + n];
}

// ---------------------------------------------------------------------------
// GEMM: out[M,128] = act( Xb[M,128] @ W[128,128] + bias )
// Xb bf16 row-major, Wt bf16 transposed [n][k]. 4 waves/block, wave = 16 rows.
// sched_group_barrier prescribes the pipeline per K-step:
//   [18 x VMEM-read (2 A + 16 B)] then [8 x WMMA back-to-back]
// so there is one load-latency exposure per K-step instead of one per tile.
// ---------------------------------------------------------------------------
__global__ __launch_bounds__(128) void gemm_bias_kernel(
    const __bf16* __restrict__ Xb, const __bf16* __restrict__ Wt,
    const float* __restrict__ bias, float* __restrict__ outF,
    __bf16* __restrict__ outB, int M, int doRelu)
{
    const int wave    = threadIdx.x >> 5;
    const int lane    = threadIdx.x & 31;
    const int halfsel = lane >> 4;        // 0: lanes 0-15, 1: lanes 16-31
    const int lrow    = lane & 15;
    const int mbase   = blockIdx.x * 64 + wave * 16;

    int arow = mbase + lrow;
    if (arow >= M) arow = M - 1;          // clamp; stores are guarded
    const __bf16* xrow = Xb + (size_t)arow * DD;

    v8f acc[8];
#pragma unroll
    for (int t = 0; t < 8; ++t) {
        float b = bias[t * 16 + lrow];
#pragma unroll
        for (int r = 0; r < 8; ++r) acc[t][r] = b;
    }

#pragma unroll
    for (int kb = 0; kb < 4; ++kb) {
        // A fragment: 16x32 bf16, K-interleave of 8 per half-wave (ISA 7.12.2)
        const int k0a = kb * 32 + halfsel * 8;
        bf16x8 alo = *(const bf16x8*)(xrow + k0a);
        bf16x8 ahi = *(const bf16x8*)(xrow + k0a + 16);
        v16bf a;
#pragma unroll
        for (int i = 0; i < 8; ++i) { a[i] = alo[i]; a[i + 8] = ahi[i]; }

        // B fragments: K-interleave of 8 per half-wave (matches A); batch all 8 tiles
        const int k0b = kb * 32 + halfsel * 8;
        v16bf bfrag[8];
#pragma unroll
        for (int t = 0; t < 8; ++t) {
            const __bf16* wrow = Wt + (size_t)(t * 16 + lrow) * DD;
            bf16x8 blo = *(const bf16x8*)(wrow + k0b);
            bf16x8 bhi = *(const bf16x8*)(wrow + k0b + 16);
#pragma unroll
            for (int i = 0; i < 8; ++i) { bfrag[t][i] = blo[i]; bfrag[t][i + 8] = bhi[i]; }
        }

#pragma unroll
        for (int t = 0; t < 8; ++t) {
            acc[t] = __builtin_amdgcn_wmma_f32_16x16x32_bf16(
                false, a, false, bfrag[t], (short)0, acc[t], false, false);
        }

        // Prescribe schedule: 18 VMEM reads, then 8 WMMA, per K-step.
        __builtin_amdgcn_sched_group_barrier(0x0020, 18, 0);  // VMEM read
        __builtin_amdgcn_sched_group_barrier(0x0008, 8, 0);   // MFMA/WMMA
    }

#pragma unroll
    for (int t = 0; t < 8; ++t) {
        int n = t * 16 + lrow;
#pragma unroll
        for (int r = 0; r < 8; ++r) {
            int m = mbase + halfsel * 8 + r;   // C/D layout: VGPR r -> M=r / 8+r
            if (m < M) {
                float v = acc[t][r];
                if (doRelu) v = v > 0.0f ? v : 0.0f;
                if (outF) outF[(size_t)m * DD + n] = v;
                if (outB) outB[(size_t)m * DD + n] = (__bf16)v;
            }
        }
    }
}

// ---------------------------------------------------------------------------
// Edge message pass: one wave per edge, lane owns 4 columns.
// agg[d] += sigmoid(k[d] + q[s] + eW^T ea + eb) * v[s]
// Edge-indexed loads (src, dst, 21 attrs) are forced wave-uniform via
// readfirstlane so they lower to scalar (SMEM) loads, keeping the vector
// memory path free for the k/q/v gathers and FP32 atomics.
// ---------------------------------------------------------------------------
__global__ __launch_bounds__(256) void edge_kernel(
    const int* __restrict__ esrc, const int* __restrict__ edst,
    const float* __restrict__ eattr, const float* __restrict__ eW,
    const float* __restrict__ eb,
    const float* __restrict__ kf, const float* __restrict__ qf,
    const float* __restrict__ vf, float* __restrict__ agg, int E)
{
    int wid  = (int)((blockIdx.x * 256 + threadIdx.x) >> 5);
    wid = __builtin_amdgcn_readfirstlane(wid);          // wave-uniform
    int lane = threadIdx.x & 31;
    if (wid >= E) return;

    const int s  = esrc[wid];                           // s_load
    const int d  = edst[wid];                           // s_load
    const int c0 = lane * 4;

    const float4 kk = *(const float4*)(kf + (size_t)d * DD + c0);
    const float4 qq = *(const float4*)(qf + (size_t)s * DD + c0);
    const float4 vv = *(const float4*)(vf + (size_t)s * DD + c0);

    float4 ew = *(const float4*)(eb + c0);
    const float* ea = eattr + (size_t)wid * 21;
#pragma unroll
    for (int j = 0; j < 21; ++j) {
        float aej = ea[j];                               // uniform -> s_load
        float4 w = *(const float4*)(eW + j * DD + c0);   // L2-hot
        ew.x += aej * w.x; ew.y += aej * w.y;
        ew.z += aej * w.z; ew.w += aej * w.w;
    }

    float4 g;
    g.x = 1.0f / (1.0f + __expf(-(kk.x + qq.x + ew.x)));
    g.y = 1.0f / (1.0f + __expf(-(kk.y + qq.y + ew.y)));
    g.z = 1.0f / (1.0f + __expf(-(kk.z + qq.z + ew.z)));
    g.w = 1.0f / (1.0f + __expf(-(kk.w + qq.w + ew.w)));

    float* ag = agg + (size_t)d * DD + c0;
    unsafeAtomicAdd(ag + 0, g.x * vv.x);
    unsafeAtomicAdd(ag + 1, g.y * vv.y);
    unsafeAtomicAdd(ag + 2, g.z * vv.z);
    unsafeAtomicAdd(ag + 3, g.w * vv.w);
}

// ---------------------------------------------------------------------------
// Column partial sums for mean pooling (coalesced rows, L2-resident)
// ---------------------------------------------------------------------------
__global__ __launch_bounds__(128) void colsum_kernel(
    const float* __restrict__ x, float* __restrict__ gsum, int M)
{
    int c  = threadIdx.x;
    int r0 = blockIdx.x * 1024;
    int r1 = r0 + 1024; if (r1 > M) r1 = M;
    float s = 0.0f;
    for (int r = r0; r < r1; ++r) s += x[(size_t)r * DD + c];
    unsafeAtomicAdd(&gsum[c], s);
}

__global__ __launch_bounds__(128) void zero128_kernel(float* p)
{
    p[threadIdx.x] = 0.0f;
}

// hb[c] = h1_b[c] + sum_k (gsum[k]/N) * h1_W[(128+k)*128 + c]
__global__ __launch_bounds__(128) void hbias_kernel(
    const float* __restrict__ gsum, const float* __restrict__ h1W,
    const float* __restrict__ h1b, float* __restrict__ hb, float invN)
{
    int c = threadIdx.x;
    float s = h1b[c];
    for (int k = 0; k < DD; ++k)
        s += gsum[k] * invN * h1W[(size_t)(DD + k) * DD + c];
    hb[c] = s;
}

// ---------------------------------------------------------------------------
// Head: out[n] = x[n,:] . w + b   (wave per node, shuffle reduction)
// ---------------------------------------------------------------------------
__global__ __launch_bounds__(256) void head3_kernel(
    const float* __restrict__ x, const float* __restrict__ w,
    const float* __restrict__ b, float* __restrict__ out, int M)
{
    int wid  = (int)((blockIdx.x * 256 + threadIdx.x) >> 5);
    int lane = threadIdx.x & 31;
    if (wid >= M) return;
    const float4 xv = *(const float4*)(x + (size_t)wid * DD + lane * 4);
    const float4 wv = *(const float4*)(w + lane * 4);
    float s = xv.x * wv.x + xv.y * wv.y + xv.z * wv.z + xv.w * wv.w;
#pragma unroll
    for (int off = 16; off > 0; off >>= 1) s += __shfl_xor(s, off, 32);
    if (lane == 0) out[wid] = s + b[0];
}

// ---------------------------------------------------------------------------
extern "C" void kernel_launch(void* const* d_in, const int* in_sizes, int n_in,
                              void* d_out, int out_size, void* d_ws, size_t ws_size,
                              hipStream_t stream)
{
    const float* G     = (const float*)d_in[0];
    const int*   eidx  = (const int*)d_in[1];
    const float* eattr = (const float*)d_in[2];
    const float* embW  = (const float*)d_in[3];
    const float* embB  = (const float*)d_in[4];
    const float* keyW  = (const float*)d_in[5];
    const float* keyB  = (const float*)d_in[6];
    const float* qryW  = (const float*)d_in[7];
    const float* qryB  = (const float*)d_in[8];
    const float* valW  = (const float*)d_in[9];
    const float* valB  = (const float*)d_in[10];
    const float* skpW  = (const float*)d_in[11];
    const float* skpB  = (const float*)d_in[12];
    const float* edgW  = (const float*)d_in[13];
    const float* edgB  = (const float*)d_in[14];
    const float* g1W   = (const float*)d_in[15];
    const float* g1B   = (const float*)d_in[16];
    const float* g2W   = (const float*)d_in[17];
    const float* g2B   = (const float*)d_in[18];
    const float* h1W   = (const float*)d_in[19];
    const float* h1B   = (const float*)d_in[20];
    const float* h2W   = (const float*)d_in[21];
    const float* h2B   = (const float*)d_in[22];
    const float* h3W   = (const float*)d_in[23];
    const float* h3B   = (const float*)d_in[24];

    const int N = in_sizes[0] / DD;
    const int E = in_sizes[1] / 2;
    const int* esrc = eidx;
    const int* edst = eidx + E;

    // ---- workspace carve-up ----
    const size_t NB = (size_t)N * DD;
    char* p = (char*)d_ws;
    __bf16* xbA  = (__bf16*)p;  p += NB * 2;
    __bf16* xbB  = (__bf16*)p;  p += NB * 2;
    float*  kf   = (float*)p;   p += NB * 4;     // also reused as fp32 x / final
    float*  qf   = (float*)p;   p += NB * 4;
    float*  vf   = (float*)p;   p += NB * 4;
    float*  agg  = (float*)p;   p += NB * 4;
    __bf16* wts  = (__bf16*)p;  p += (size_t)15 * DD * DD * 2;
    float*  gsum = (float*)p;   p += 128 * 4;
    float*  hb   = (float*)p;   p += 128 * 4;
    (void)ws_size; (void)n_in; (void)out_size;

    const size_t WSZ = (size_t)DD * DD;
    __bf16* wt_emb0 = wts + 0 * WSZ;
    __bf16* wt_emb1 = wts + 1 * WSZ;
    __bf16* wt_emb2 = wts + 2 * WSZ;
    __bf16* wt_key0 = wts + 3 * WSZ;
    __bf16* wt_key1 = wts + 4 * WSZ;
    __bf16* wt_qry0 = wts + 5 * WSZ;
    __bf16* wt_qry1 = wts + 6 * WSZ;
    __bf16* wt_val0 = wts + 7 * WSZ;
    __bf16* wt_val1 = wts + 8 * WSZ;
    __bf16* wt_skp0 = wts + 9 * WSZ;
    __bf16* wt_skp1 = wts + 10 * WSZ;
    __bf16* wt_g1   = wts + 11 * WSZ;
    __bf16* wt_g2   = wts + 12 * WSZ;
    __bf16* wt_h1   = wts + 13 * WSZ;   // top 128 rows of h1_W
    __bf16* wt_h2   = wts + 14 * WSZ;

    // ---- weight prep (bf16 transpose) ----
    wtrans_kernel<<<64, 256, 0, stream>>>(embW + 0 * WSZ, wt_emb0);
    wtrans_kernel<<<64, 256, 0, stream>>>(embW + 1 * WSZ, wt_emb1);
    wtrans_kernel<<<64, 256, 0, stream>>>(embW + 2 * WSZ, wt_emb2);
    wtrans_kernel<<<64, 256, 0, stream>>>(keyW + 0 * WSZ, wt_key0);
    wtrans_kernel<<<64, 256, 0, stream>>>(keyW + 1 * WSZ, wt_key1);
    wtrans_kernel<<<64, 256, 0, stream>>>(qryW + 0 * WSZ, wt_qry0);
    wtrans_kernel<<<64, 256, 0, stream>>>(qryW + 1 * WSZ, wt_qry1);
    wtrans_kernel<<<64, 256, 0, stream>>>(valW + 0 * WSZ, wt_val0);
    wtrans_kernel<<<64, 256, 0, stream>>>(valW + 1 * WSZ, wt_val1);
    wtrans_kernel<<<64, 256, 0, stream>>>(skpW + 0 * WSZ, wt_skp0);
    wtrans_kernel<<<64, 256, 0, stream>>>(skpW + 1 * WSZ, wt_skp1);
    wtrans_kernel<<<64, 256, 0, stream>>>(g1W, wt_g1);
    wtrans_kernel<<<64, 256, 0, stream>>>(g2W, wt_g2);
    wtrans_kernel<<<64, 256, 0, stream>>>(h1W, wt_h1);   // rows 0..127
    wtrans_kernel<<<64, 256, 0, stream>>>(h2W, wt_h2);

    const int cvtGrid  = (int)((NB + 255) / 256);
    const int gemmGrid = (N + 63) / 64;
    const int edgeGrid = (E + 7) / 8;        // 8 waves/block, wave per edge
    const int csGrid   = (N + 1023) / 1024;
    const int hGrid    = (N + 7) / 8;        // head3: 8 waves/block

    // ---- embedding MLP (3x GEMM+ReLU) ----
    cvt_bf16_kernel<<<cvtGrid, 256, 0, stream>>>(G, xbA, (long long)NB, 0);
    gemm_bias_kernel<<<gemmGrid, 128, 0, stream>>>(xbA, wt_emb0, embB + 0,   nullptr, xbB, N, 1);
    gemm_bias_kernel<<<gemmGrid, 128, 0, stream>>>(xbB, wt_emb1, embB + 128, nullptr, xbA, N, 1);
    gemm_bias_kernel<<<gemmGrid, 128, 0, stream>>>(xbA, wt_emb2, embB + 256, nullptr, xbB, N, 1);

    // ---- conv layer 0 (x in xbB) ----
    gemm_bias_kernel<<<gemmGrid, 128, 0, stream>>>(xbB, wt_key0, keyB + 0, kf,  nullptr, N, 0);
    gemm_bias_kernel<<<gemmGrid, 128, 0, stream>>>(xbB, wt_qry0, qryB + 0, qf,  nullptr, N, 0);
    gemm_bias_kernel<<<gemmGrid, 128, 0, stream>>>(xbB, wt_val0, valB + 0, vf,  nullptr, N, 0);
    gemm_bias_kernel<<<gemmGrid, 128, 0, stream>>>(xbB, wt_skp0, skpB + 0, agg, nullptr, N, 0); // agg = skip
    edge_kernel<<<edgeGrid, 256, 0, stream>>>(esrc, edst, eattr,
        edgW + 0, edgB + 0, kf, qf, vf, agg, E);
    cvt_bf16_kernel<<<cvtGrid, 256, 0, stream>>>(agg, xbA, (long long)NB, 1);

    // ---- conv layer 1 (x in xbA) ----
    gemm_bias_kernel<<<gemmGrid, 128, 0, stream>>>(xbA, wt_key1, keyB + 128, kf,  nullptr, N, 0);
    gemm_bias_kernel<<<gemmGrid, 128, 0, stream>>>(xbA, wt_qry1, qryB + 128, qf,  nullptr, N, 0);
    gemm_bias_kernel<<<gemmGrid, 128, 0, stream>>>(xbA, wt_val1, valB + 128, vf,  nullptr, N, 0);
    gemm_bias_kernel<<<gemmGrid, 128, 0, stream>>>(xbA, wt_skp1, skpB + 128, agg, nullptr, N, 0);
    edge_kernel<<<edgeGrid, 256, 0, stream>>>(esrc, edst, eattr,
        edgW + 21 * DD, edgB + 128, kf, qf, vf, agg, E);
    cvt_bf16_kernel<<<cvtGrid, 256, 0, stream>>>(agg, xbB, (long long)NB, 1);

    // ---- graph linears ----
    gemm_bias_kernel<<<gemmGrid, 128, 0, stream>>>(xbB, wt_g1, g1B, nullptr, xbA, N, 1);
    gemm_bias_kernel<<<gemmGrid, 128, 0, stream>>>(xbA, wt_g2, g2B, kf, xbB, N, 0); // kf = fp32 x

    // ---- mean pooling + fold gmean half of h1 into bias ----
    zero128_kernel<<<1, 128, 0, stream>>>(gsum);
    colsum_kernel<<<csGrid, 128, 0, stream>>>(kf, gsum, N);
    hbias_kernel<<<1, 128, 0, stream>>>(gsum, h1W, h1B, hb, 1.0f / (float)N);

    // ---- head ----
    gemm_bias_kernel<<<gemmGrid, 128, 0, stream>>>(xbB, wt_h1, hb,  nullptr, xbA, N, 1);
    gemm_bias_kernel<<<gemmGrid, 128, 0, stream>>>(xbA, wt_h2, h2B, kf, nullptr, N, 1);
    head3_kernel<<<hGrid, 256, 0, stream>>>(kf, h3W, h3B, (float*)d_out, N);
}
